// BigramModel_39651138076765
// MI455X (gfx1250) — matrix operand order repaired and checked by reading, not compiled, not run on hardware
//
#include <hip/hip_runtime.h>
#include <hip/hip_bf16.h>

typedef _Float16 half_t;
typedef __attribute__((ext_vector_type(16))) _Float16 v16h;
typedef __attribute__((ext_vector_type(8)))  _Float16 v8h;
typedef __attribute__((ext_vector_type(8)))  float    v8f;
typedef __attribute__((ext_vector_type(4)))  int      v4i;

#define NB   8
#define TT   1024
#define DD   384
#define NH   6
#define HSZ  64
#define FFD  1536
#define NV   32000
#define NBT  (NB * TT)
#define NLAYER 3

#define HAS_ASYNC_LDS __has_builtin(__builtin_amdgcn_global_load_async_to_lds_b128)

// ---------------------------------------------------------------------------
// WMMA fragment loaders (CDNA5 16x16x32 f16 layouts, CDNA5 ISA)
// ---------------------------------------------------------------------------
union F16Frag { v16h v; v8h h[2]; };

__device__ __forceinline__ v16h load_a_frag(const half_t* tile, int ld, int lane) {
  const int m   = lane & 15;
  const int off = (lane >> 4) << 3;          // 0 or 8
  const half_t* p = tile + (size_t)m * ld + off;
  F16Frag u;
  u.h[0] = *(const v8h*)(p);
  u.h[1] = *(const v8h*)(p + 16);
  return u.v;
}

__device__ __forceinline__ v16h load_b_frag(const half_t* tile, int ld, int lane) {
  const int n   = lane & 15;
  const int off = (lane >> 4) << 3;          // 0 or 8
  const half_t* p = tile + (size_t)n * ld + off;
  F16Frag u;
  u.h[0] = *(const v8h*)(p);
  u.h[1] = *(const v8h*)(p + 16);
  return u.v;
}

__device__ __forceinline__ v8f wmma_f16(v16h a, v16h b, v8f c) {
  return __builtin_amdgcn_wmma_f32_16x16x32_f16(false, a, false, b, (short)0, c,
                                                false, false);
}

// Stage 16 bytes global -> LDS. Prefer the CDNA5 async path (ASYNCcnt DMA,
// no VGPR round-trip); fall back to a synchronous reg-staged copy.
__device__ __forceinline__ void stage_b128(const half_t* gsrc, half_t* lds_dst) {
#if HAS_ASYNC_LDS
  __builtin_amdgcn_global_load_async_to_lds_b128(
      (__attribute__((address_space(1))) v4i*)gsrc,
      (__attribute__((address_space(3))) v4i*)lds_dst, 0, 0);
#else
  *(v8h*)lds_dst = *(const v8h*)gsrc;
#endif
}

__device__ __forceinline__ void stage_wait() {
#if HAS_ASYNC_LDS
#if __has_builtin(__builtin_amdgcn_s_wait_asynccnt)
  __builtin_amdgcn_s_wait_asynccnt(0);
#else
  asm volatile("s_wait_asynccnt 0x0" ::: "memory");
#endif
#endif
}

// ---------------------------------------------------------------------------
// Embedding: x[b,t,:] = tok[idx[b,t],:] + pos[t,:]
// ---------------------------------------------------------------------------
__global__ __launch_bounds__(256)
void embed_kernel(const int* __restrict__ idxs, const float* __restrict__ tok,
                  const float* __restrict__ pos, float* __restrict__ x) {
  int i = blockIdx.x * 256 + threadIdx.x;
  if (i >= NBT * DD) return;
  int bt = i / DD, d = i - bt * DD;
  int t = bt & (TT - 1);
  x[i] = tok[(size_t)idxs[bt] * DD + d] + pos[(size_t)t * DD + d];
}

// ---------------------------------------------------------------------------
// LayerNorm -> f16, one wave per row (wave32 shfl reductions)
// ---------------------------------------------------------------------------
__global__ __launch_bounds__(32)
void layernorm_f16_kernel(const float* __restrict__ x, const float* __restrict__ g,
                          const float* __restrict__ b, half_t* __restrict__ out) {
  int row = blockIdx.x;
  int lane = threadIdx.x;
  const float* xr = x + (size_t)row * DD;
  float s = 0.f, ss = 0.f;
  for (int i = lane; i < DD; i += 32) { float v = xr[i]; s += v; ss += v * v; }
  #pragma unroll
  for (int m = 16; m > 0; m >>= 1) { s += __shfl_xor(s, m, 32); ss += __shfl_xor(ss, m, 32); }
  float mu = s * (1.f / DD);
  float var = ss * (1.f / DD) - mu * mu;
  float rstd = rsqrtf(var + 1e-5f);
  half_t* orow = out + (size_t)row * DD;
  for (int i = lane; i < DD; i += 32)
    orow[i] = (half_t)((xr[i] - mu) * rstd * g[i] + b[i]);
}

// ---------------------------------------------------------------------------
// fp32 [K,N] -> f16 [N,K] transpose-convert (batched via blockIdx.y)
// ---------------------------------------------------------------------------
__global__ __launch_bounds__(256)
void transpose_to_f16_kernel(const float* __restrict__ src, half_t* __restrict__ dst,
                             int K, int N) {
  size_t base = (size_t)blockIdx.y * K * N;
  int i = blockIdx.x * 256 + threadIdx.x;
  if (i >= K * N) return;
  int n = i / K, k = i - n * K;
  dst[base + i] = (half_t)src[base + (size_t)k * N + n];
}

// ---------------------------------------------------------------------------
// WMMA GEMM: C[M,N] = A[M,K](f16,row) * Bt[N,K](f16,row) [+bias][+resid][relu]
// 8 waves/block; wave tile 32x64 (2 A-frags x 4 B-frags, 8 accumulators);
// block tile 256x64. The 64x32 B tile is staged into double-buffered LDS via
// async global->LDS copies (one 4KB tile per block per k-step), k-loop
// unrolled by two with ping-pong register/LDS sets. Requires K % 64 == 0.
// mode: 0 = f32 plain, 1 = f16 plain, 2 = f16 per-head [B,H,T,HS] scatter,
//       3 = f16 per-head transposed [B,H,HS,T] scatter (for V)
// ---------------------------------------------------------------------------
__global__ __launch_bounds__(256)
void gemm_wmma_kernel(const half_t* __restrict__ A, const half_t* __restrict__ Bt,
                      const float* __restrict__ bias, const float* __restrict__ resid,
                      float* __restrict__ outF, half_t* __restrict__ outH,
                      int M, int N, int K, int mode, int relu) {
  __shared__ half_t Bs0[64 * 32];
  __shared__ half_t Bs1[64 * 32];
  const int tid  = threadIdx.x;
  const int wave = tid >> 5;
  const int lane = tid & 31;
  const int m0 = blockIdx.y * 256 + wave * 32;
  const int n0 = blockIdx.x * 64;

  // B staging: thread -> (row = tid/4, 8-half segment = tid%4)
  const int srow = tid >> 2;
  const int scol = (tid & 3) << 3;
  const half_t* Bsrc = Bt + (size_t)(n0 + srow) * K + scol;
  half_t* Bdst0 = &Bs0[srow * 32 + scol];
  half_t* Bdst1 = &Bs1[srow * 32 + scol];

  v8f acc[2][4];
  #pragma unroll
  for (int mt = 0; mt < 2; ++mt)
    #pragma unroll
    for (int nt = 0; nt < 4; ++nt) acc[mt][nt] = v8f{};

  const half_t* A0 = A + (size_t)m0 * K;
  const half_t* A1 = A0 + (size_t)16 * K;

  // prologue: k-step 0 into Bs0 / aP
  stage_b128(Bsrc, Bdst0);
  v16h a0P = load_a_frag(A0, K, lane);
  v16h a1P = load_a_frag(A1, K, lane);
  stage_wait();
  __syncthreads();

  for (int k0 = 0; k0 < K; k0 += 64) {
    // ---- half-step A: stage k0+32 -> Bs1 / aQ; compute on (Bs0, aP)
    stage_b128(Bsrc + k0 + 32, Bdst1);
    v16h a0Q = load_a_frag(A0 + k0 + 32, K, lane);
    v16h a1Q = load_a_frag(A1 + k0 + 32, K, lane);
    {
      v16h cb[4];
      #pragma unroll
      for (int nt = 0; nt < 4; ++nt) cb[nt] = load_b_frag(&Bs0[(nt * 16) * 32], 32, lane);
      #pragma unroll
      for (int nt = 0; nt < 4; ++nt) acc[0][nt] = wmma_f16(a0P, cb[nt], acc[0][nt]);
      #pragma unroll
      for (int nt = 0; nt < 4; ++nt) acc[1][nt] = wmma_f16(a1P, cb[nt], acc[1][nt]);
    }
    stage_wait();
    __syncthreads();

    // ---- half-step B: stage k0+64 -> Bs0 / aP; compute on (Bs1, aQ)
    const bool more = (k0 + 64) < K;
    if (more) {
      stage_b128(Bsrc + k0 + 64, Bdst0);
      a0P = load_a_frag(A0 + k0 + 64, K, lane);
      a1P = load_a_frag(A1 + k0 + 64, K, lane);
    }
    {
      v16h cb[4];
      #pragma unroll
      for (int nt = 0; nt < 4; ++nt) cb[nt] = load_b_frag(&Bs1[(nt * 16) * 32], 32, lane);
      #pragma unroll
      for (int nt = 0; nt < 4; ++nt) acc[0][nt] = wmma_f16(a0Q, cb[nt], acc[0][nt]);
      #pragma unroll
      for (int nt = 0; nt < 4; ++nt) acc[1][nt] = wmma_f16(a1Q, cb[nt], acc[1][nt]);
    }
    stage_wait();
    __syncthreads();
  }

  const int nn = lane & 15;
  const int rb = (lane >> 4) << 3;
  #pragma unroll
  for (int mt = 0; mt < 2; ++mt) {
    #pragma unroll
    for (int nt = 0; nt < 4; ++nt) {
      #pragma unroll
      for (int r = 0; r < 8; ++r) {
        int m = m0 + mt * 16 + rb + r;
        int n = n0 + nt * 16 + nn;
        float v = acc[mt][nt][r];
        if (bias)  v += bias[n];
        if (resid) v += resid[(size_t)m * N + n];
        if (relu)  v = fmaxf(v, 0.f);
        if (mode == 0) {
          outF[(size_t)m * N + n] = v;
        } else if (mode == 1) {
          outH[(size_t)m * N + n] = (half_t)v;
        } else {
          int b = m / TT, t = m - b * TT;
          int h = n >> 6, s = n & 63;
          if (mode == 2)
            outH[(((size_t)(b * NH + h)) * TT + t) * HSZ + s] = (half_t)v;
          else
            outH[(((size_t)(b * NH + h)) * HSZ + s) * TT + t] = (half_t)v;
        }
      }
    }
  }
}

// ---------------------------------------------------------------------------
// Flash attention: one wave per (b,h, 16-query tile). Q,K: [B,H,T,HS] f16,
// Vt: [B,H,HS,T] f16. Output written f16 into concat layout [B,T, h*64+s].
// ---------------------------------------------------------------------------
__global__ __launch_bounds__(32)
void attn_kernel(const half_t* __restrict__ Q, const half_t* __restrict__ Km,
                 const half_t* __restrict__ Vt, half_t* __restrict__ out) {
  __shared__ half_t P[16 * 32];
  const int qt = blockIdx.x;            // query tile (T/16)
  const int bh = blockIdx.y;            // b*H + h
  const int lane = threadIdx.x;
  const int b = bh / NH, h = bh - b * NH;
  const int qbase = qt * 16;
  const half_t* Qb = Q  + (size_t)bh * TT * HSZ;
  const half_t* Kb = Km + (size_t)bh * TT * HSZ;
  const half_t* Vb = Vt + (size_t)bh * HSZ * TT;

  v16h qf0 = load_a_frag(Qb + (size_t)qbase * HSZ +  0, HSZ, lane);
  v16h qf1 = load_a_frag(Qb + (size_t)qbase * HSZ + 32, HSZ, lane);

  const int nn = lane & 15;
  const int rb = (lane >> 4) << 3;
  float mrow[8], lrow[8];
  #pragma unroll
  for (int r = 0; r < 8; ++r) { mrow[r] = -__builtin_inff(); lrow[r] = 0.f; }
  v8f o[4] = {v8f{}, v8f{}, v8f{}, v8f{}};

  const float scale = 0.125f;           // HS^-0.5
  const int nblocks = (qbase + 15) / 32 + 1;
  for (int j = 0; j < nblocks; ++j) {
    const int kb = j * 32;
    v16h kf0 = load_b_frag(Kb + (size_t)kb * HSZ +  0, HSZ, lane);
    v16h kf1 = load_b_frag(Kb + (size_t)kb * HSZ + 32, HSZ, lane);
    v16h kf2 = load_b_frag(Kb + (size_t)(kb + 16) * HSZ +  0, HSZ, lane);
    v16h kf3 = load_b_frag(Kb + (size_t)(kb + 16) * HSZ + 32, HSZ, lane);
    v8f s0 = {}, s1 = {};
    s0 = wmma_f16(qf0, kf0, s0);
    s0 = wmma_f16(qf1, kf1, s0);
    s1 = wmma_f16(qf0, kf2, s1);
    s1 = wmma_f16(qf1, kf3, s1);

    // prefetch V fragments now; softmax below hides the latency
    v16h vf[4];
    #pragma unroll
    for (int nt = 0; nt < 4; ++nt)
      vf[nt] = load_b_frag(Vb + (size_t)(nt * 16) * TT + kb, TT, lane);

    #pragma unroll
    for (int r = 0; r < 8; ++r) {
      const int row = qbase + rb + r;
      float a0 = s0[r] * scale;
      float a1 = s1[r] * scale;
      if (kb + nn > row)      a0 = -__builtin_inff();
      if (kb + 16 + nn > row) a1 = -__builtin_inff();
      float mx = fmaxf(a0, a1);
      #pragma unroll
      for (int msk = 8; msk > 0; msk >>= 1) mx = fmaxf(mx, __shfl_xor(mx, msk, 32));
      float mnew = fmaxf(mrow[r], mx);
      float sc = __expf(mrow[r] - mnew);
      mrow[r] = mnew;
      float e0 = __expf(a0 - mnew);
      float e1 = __expf(a1 - mnew);
      float rs = e0 + e1;
      #pragma unroll
      for (int msk = 8; msk > 0; msk >>= 1) rs += __shfl_xor(rs, msk, 32);
      lrow[r] = lrow[r] * sc + rs;
      #pragma unroll
      for (int nt = 0; nt < 4; ++nt) o[nt][r] *= sc;
      P[(rb + r) * 32 + nn]      = (half_t)e0;
      P[(rb + r) * 32 + nn + 16] = (half_t)e1;
    }
    __syncthreads();
    v16h pf = load_a_frag(P, 32, lane);
    #pragma unroll
    for (int nt = 0; nt < 4; ++nt)
      o[nt] = wmma_f16(pf, vf[nt], o[nt]);
    __syncthreads();
  }

  float rinv[8];
  #pragma unroll
  for (int r = 0; r < 8; ++r) rinv[r] = 1.f / lrow[r];
  #pragma unroll
  for (int nt = 0; nt < 4; ++nt) {
    #pragma unroll
    for (int r = 0; r < 8; ++r) {
      int m = qbase + rb + r;
      out[((size_t)(b * TT + m)) * DD + h * HSZ + nt * 16 + nn] = (half_t)(o[nt][r] * rinv[r]);
    }
  }
}

// ---------------------------------------------------------------------------
// Cross-entropy: per-row logsumexp - target logit, atomic accumulate
// ---------------------------------------------------------------------------
__global__ __launch_bounds__(256)
void loss_row_kernel(const float* __restrict__ logits, const int* __restrict__ targets,
                     float* __restrict__ accum) {
  __shared__ float red[256];
  const int row = blockIdx.x;
  const int tid = threadIdx.x;
  const float* lr = logits + (size_t)row * NV;
  float mx = -__builtin_inff();
  for (int i = tid; i < NV; i += 256) mx = fmaxf(mx, lr[i]);
  red[tid] = mx; __syncthreads();
  for (int s = 128; s > 0; s >>= 1) { if (tid < s) red[tid] = fmaxf(red[tid], red[tid + s]); __syncthreads(); }
  mx = red[0]; __syncthreads();
  float sum = 0.f;
  for (int i = tid; i < NV; i += 256) sum += __expf(lr[i] - mx);
  red[tid] = sum; __syncthreads();
  for (int s = 128; s > 0; s >>= 1) { if (tid < s) red[tid] += red[tid + s]; __syncthreads(); }
  if (tid == 0) {
    float lse = mx + __logf(red[0]);
    atomicAdd(accum, lse - lr[targets[row]]);
  }
}

__global__ void zero1_kernel(float* p) { p[0] = 0.f; }
__global__ void loss_final_kernel(const float* accum, float* loss_out) {
  loss_out[0] = accum[0] * (1.f / NBT);
}

// ---------------------------------------------------------------------------
// Host orchestration
// ---------------------------------------------------------------------------
static char* bump(char*& p, size_t bytes) {
  char* r = p;
  p += (bytes + 255) & ~(size_t)255;
  return r;
}

extern "C" void kernel_launch(void* const* d_in, const int* in_sizes, int n_in,
                              void* d_out, int out_size, void* d_ws, size_t ws_size,
                              hipStream_t stream) {
  (void)in_sizes; (void)n_in; (void)out_size; (void)ws_size;
  const int*   idxs    = (const int*)d_in[0];
  const int*   targets = (const int*)d_in[1];
  const float* tok     = (const float*)d_in[2];
  const float* pos     = (const float*)d_in[3];
  // blocks: base 4 + L*12 : wq wk wv wo w1 b1 w2 b2 g1 be1 g2 be2
  const float* gf  = (const float*)d_in[4 + NLAYER * 12 + 0];
  const float* bf  = (const float*)d_in[4 + NLAYER * 12 + 1];
  const float* wlm = (const float*)d_in[4 + NLAYER * 12 + 2];
  const float* blm = (const float*)d_in[4 + NLAYER * 12 + 3];

  float* logits = (float*)d_out;
  float* loss_out = logits + (size_t)NBT * NV;

  char* wp = (char*)d_ws;
  float*  x     = (float*)bump(wp, (size_t)NBT * DD * 4);
  half_t* h16   = (half_t*)bump(wp, (size_t)NBT * DD * 2);
  half_t* q16   = (half_t*)bump(wp, (size_t)NBT * DD * 2);
  half_t* k16   = (half_t*)bump(wp, (size_t)NBT * DD * 2);
  half_t* v16   = (half_t*)bump(wp, (size_t)NBT * DD * 2);
  half_t* att16 = (half_t*)bump(wp, (size_t)NBT * DD * 2);
  half_t* hid16 = (half_t*)bump(wp, (size_t)NBT * FFD * 2);
  half_t* wq_t  = (half_t*)bump(wp, (size_t)DD * DD * 2);
  half_t* wk_t  = (half_t*)bump(wp, (size_t)DD * DD * 2);
  half_t* wv_t  = (half_t*)bump(wp, (size_t)DD * DD * 2);
  half_t* wo_t  = (half_t*)bump(wp, (size_t)DD * DD * 2);
  half_t* w1_t  = (half_t*)bump(wp, (size_t)DD * FFD * 2);
  half_t* w2_t  = (half_t*)bump(wp, (size_t)FFD * DD * 2);
  half_t* wlm_t = (half_t*)bump(wp, (size_t)DD * NV * 2);
  float*  accum = (float*)bump(wp, 256);

  // LM head weight transpose (done once per call)
  {
    int elems = DD * NV;
    transpose_to_f16_kernel<<<dim3((elems + 255) / 256, 1), 256, 0, stream>>>(wlm, wlm_t, DD, NV);
  }

  // Embedding
  embed_kernel<<<(NBT * DD + 255) / 256, 256, 0, stream>>>(idxs, tok, pos, x);

  for (int L = 0; L < NLAYER; ++L) {
    const float* wq  = (const float*)d_in[4 + L * 12 + 0];
    const float* wk  = (const float*)d_in[4 + L * 12 + 1];
    const float* wv  = (const float*)d_in[4 + L * 12 + 2];
    const float* wo  = (const float*)d_in[4 + L * 12 + 3];
    const float* w1  = (const float*)d_in[4 + L * 12 + 4];
    const float* b1  = (const float*)d_in[4 + L * 12 + 5];
    const float* w2  = (const float*)d_in[4 + L * 12 + 6];
    const float* b2  = (const float*)d_in[4 + L * 12 + 7];
    const float* g1  = (const float*)d_in[4 + L * 12 + 8];
    const float* be1 = (const float*)d_in[4 + L * 12 + 9];
    const float* g2  = (const float*)d_in[4 + L * 12 + 10];
    const float* be2 = (const float*)d_in[4 + L * 12 + 11];

    // weight conversion: per-head [H,D,HS] -> [H*HS, D]; others [K,N] -> [N,K]
    transpose_to_f16_kernel<<<dim3((DD * HSZ + 255) / 256, NH), 256, 0, stream>>>(wq, wq_t, DD, HSZ);
    transpose_to_f16_kernel<<<dim3((DD * HSZ + 255) / 256, NH), 256, 0, stream>>>(wk, wk_t, DD, HSZ);
    transpose_to_f16_kernel<<<dim3((DD * HSZ + 255) / 256, NH), 256, 0, stream>>>(wv, wv_t, DD, HSZ);
    transpose_to_f16_kernel<<<dim3((DD * DD + 255) / 256, 1), 256, 0, stream>>>(wo, wo_t, DD, DD);
    transpose_to_f16_kernel<<<dim3((DD * FFD + 255) / 256, 1), 256, 0, stream>>>(w1, w1_t, DD, FFD);
    transpose_to_f16_kernel<<<dim3((FFD * DD + 255) / 256, 1), 256, 0, stream>>>(w2, w2_t, FFD, DD);

    // LN1 -> h16
    layernorm_f16_kernel<<<NBT, 32, 0, stream>>>(x, g1, be1, h16);

    // QKV projections (block tile 256x64)
    dim3 gqkv(DD / 64, NBT / 256);
    gemm_wmma_kernel<<<gqkv, 256, 0, stream>>>(h16, wq_t, nullptr, nullptr, nullptr, q16,
                                               NBT, DD, DD, 2, 0);
    gemm_wmma_kernel<<<gqkv, 256, 0, stream>>>(h16, wk_t, nullptr, nullptr, nullptr, k16,
                                               NBT, DD, DD, 2, 0);
    gemm_wmma_kernel<<<gqkv, 256, 0, stream>>>(h16, wv_t, nullptr, nullptr, nullptr, v16,
                                               NBT, DD, DD, 3, 0);

    // attention
    attn_kernel<<<dim3(TT / 16, NB * NH), 32, 0, stream>>>(q16, k16, v16, att16);

    // x = x + att @ wo
    gemm_wmma_kernel<<<dim3(DD / 64, NBT / 256), 256, 0, stream>>>(
        att16, wo_t, nullptr, x, x, nullptr, NBT, DD, DD, 0, 0);

    // LN2 -> h16
    layernorm_f16_kernel<<<NBT, 32, 0, stream>>>(x, g2, be2, h16);

    // MLP
    gemm_wmma_kernel<<<dim3(FFD / 64, NBT / 256), 256, 0, stream>>>(
        h16, w1_t, b1, nullptr, nullptr, hid16, NBT, FFD, DD, 1, 1);
    gemm_wmma_kernel<<<dim3(DD / 64, NBT / 256), 256, 0, stream>>>(
        hid16, w2_t, b2, x, x, nullptr, NBT, DD, FFD, 0, 0);
  }

  // final LN -> h16, then LM head directly into d_out
  layernorm_f16_kernel<<<NBT, 32, 0, stream>>>(x, gf, bf, h16);
  gemm_wmma_kernel<<<dim3(NV / 64, NBT / 256), 256, 0, stream>>>(
      h16, wlm_t, blm, nullptr, logits, nullptr, NBT, NV, DD, 0, 0);

  // loss
  zero1_kernel<<<1, 1, 0, stream>>>(accum);
  loss_row_kernel<<<NBT, 256, 0, stream>>>(logits, targets, accum);
  loss_final_kernel<<<1, 1, 0, stream>>>(accum, loss_out);
}
